// WindowAttention_609885356485
// MI455X (gfx1250) — compile-verified, matrix operand order checked by reading, not run on hardware
//
#include <hip/hip_runtime.h>

// =====================================================================
// MI455X (gfx1250, CDNA5) fused Swin window attention.
// One workgroup (256 thr = 8 wave32) per window: TDM tensor_load_to_lds
// stages X, then QKV GEMM -> 12-head attention (softmax w/ relative
// position bias) -> projection GEMM, all intermediates in LDS.
// Matrix math via v_wmma_f32_16x16x32_bf16 (f32 accumulate).
// =====================================================================

typedef __attribute__((ext_vector_type(16))) __bf16 v16bf;
typedef __attribute__((ext_vector_type(8)))  __bf16 v8bf;
typedef __attribute__((ext_vector_type(8)))  float  v8f;
typedef __attribute__((ext_vector_type(4)))  unsigned int v4u;
typedef __attribute__((ext_vector_type(8)))  int v8i;
typedef __attribute__((ext_vector_type(4)))  int v4i;

#define N_TOK   49
#define N_PAD   64
#define N_HEADS 12
#define HEAD_D  32
#define EMB     384
#define SS_LD   66   // padded f32 stride for score slice (bank-conflict break)
#define PS_LD   72   // padded bf16 stride for P slice (multiple of 8 -> b128 ok)

// ---- dynamic LDS layout (bytes) ----
#define XS_OFF   0                            // [64][384] bf16: X, later attn-out
#define QKS_OFF  (XS_OFF + 64 * 384 * 2)      // [64][768] bf16: Q|K (also f32 X staging)
#define VTS_OFF  (QKS_OFF + 64 * 768 * 2)     // [384][64] bf16: V transposed (row = h*32+d)
#define SS_OFF   (VTS_OFF + 384 * 64 * 2)     // 8 waves * [16][SS_LD] f32
#define PS_OFF   (SS_OFF + 8 * 16 * SS_LD * 4)// 8 waves * [16][PS_LD] bf16
#define LDS_BYTES (PS_OFF + 8 * 16 * PS_LD * 2)   // = 248832

__device__ __forceinline__ v8f wmma_bf16(v16bf a, v16bf b, v8f c) {
  return __builtin_amdgcn_wmma_f32_16x16x32_bf16(
      /*neg_a=*/false, a, /*neg_b=*/false, b,
      /*c_mod=*/(short)0, c, /*reuse_a=*/false, /*reuse_b=*/false);
}

// Load a 16x32 A-fragment (or symmetric 32x16 B-fragment from an [N][K]
// "column-major B" array): lane<16 holds row (lane&15), K {0..7,16..23};
// lane>=16 holds K {8..15,24..31}. Two contiguous 16B chunks per lane.
__device__ __forceinline__ v16bf load_frag(const __bf16* __restrict__ base,
                                           int ld, int row_base, int k_base,
                                           int lane) {
  const int r  = row_base + (lane & 15);
  const int kc = k_base + ((lane >> 4) << 3);
  const __bf16* p = base + r * ld + kc;
  v8bf c0 = *(const v8bf*)(p);
  v8bf c1 = *(const v8bf*)(p + 16);
  v16bf f;
#pragma unroll
  for (int i = 0; i < 8; ++i) { f[i] = c0[i]; f[i + 8] = c1[i]; }
  return f;
}

// C/D layout: VGPR r -> row Mb + (lane>=16 ? 8 : 0) + r, col Nb + (lane&15).
__device__ __forceinline__ void store_qkv_acc(v8f acc, int Mb, int Nb, int lane,
                                              __bf16* qks, __bf16* vts, float bias) {
  const int lcol = lane & 15;
  const int lh8  = (lane >> 4) << 3;
  const int j    = Nb + lcol;
  if (Nb < 768) {                       // Q|K region: column store into [64][768]
#pragma unroll
    for (int r = 0; r < 8; ++r)
      qks[(Mb + lh8 + r) * 768 + j] = (__bf16)(acc[r] + bias);
  } else {                              // V region: transposed store -> one b128
    const int d = j - 768;
    v8bf vv;
#pragma unroll
    for (int r = 0; r < 8; ++r) vv[r] = (__bf16)(acc[r] + bias);
    *(v8bf*)(vts + d * 64 + Mb + lh8) = vv;
  }
}

// ---------------- prep: weights f32->bf16, expand relative-position bias ----
__global__ void prep_kernel(const float* __restrict__ qkv_w,
                            const float* __restrict__ proj_w,
                            const float* __restrict__ rbt,
                            unsigned short* __restrict__ qkvw_u,
                            unsigned short* __restrict__ projw_u,
                            float* __restrict__ bias_exp) {
  __bf16* qkvw  = (__bf16*)qkvw_u;
  __bf16* projw = (__bf16*)projw_u;
  const int i = blockIdx.x * blockDim.x + threadIdx.x;
  const int stride = gridDim.x * blockDim.x;
  for (int t = i; t < 1152 * 384; t += stride) qkvw[t]  = (__bf16)qkv_w[t];
  for (int t = i; t < 384 * 384;  t += stride) projw[t] = (__bf16)proj_w[t];
  for (int t = i; t < N_HEADS * N_TOK * N_TOK; t += stride) {
    const int h = t / (N_TOK * N_TOK);
    const int rem = t - h * N_TOK * N_TOK;
    const int n = rem / N_TOK, m = rem % N_TOK;
    const int idx = ((n / 7) - (m / 7) + 6) * 13 + ((n % 7) - (m % 7) + 6);
    bias_exp[t] = rbt[idx * N_HEADS + h];
  }
}

// ---------------- fused per-window kernel ----------------
__global__ __launch_bounds__(256, 1)
void swin_attn_kernel(const float* __restrict__ x,
                      const unsigned short* __restrict__ qkvw_u,
                      const float* __restrict__ qkvb,
                      const unsigned short* __restrict__ projw_u,
                      const float* __restrict__ projb,
                      const float* __restrict__ bias_exp,
                      float* __restrict__ out) {
  extern __shared__ char smem[];
  __bf16* xs  = (__bf16*)(smem + XS_OFF);
  __bf16* qks = (__bf16*)(smem + QKS_OFF);
  __bf16* vts = (__bf16*)(smem + VTS_OFF);
  float*  ss  = (float*) (smem + SS_OFF);
  __bf16* ps  = (__bf16*)(smem + PS_OFF);
  const __bf16* qkvw  = (const __bf16*)qkvw_u;
  const __bf16* projw = (const __bf16*)projw_u;

  const int b    = blockIdx.x;
  const int tid  = threadIdx.x;
  const int wave = tid >> 5;
  const int lane = tid & 31;
  const int lcol = lane & 15;
  const int lh8  = (lane >> 4) << 3;

  const float* xw = x + (size_t)b * N_TOK * EMB;

#if __has_builtin(__builtin_amdgcn_tensor_load_to_lds) && __has_builtin(__builtin_amdgcn_s_wait_tensorcnt)
  // -------- Phase 0a: TDM DMA of the 49x384 f32 X tile into LDS staging ----
  // D# group0: count=1 | lds_addr | global_addr[56:0] | type=2.
  // D# group1: data_size=4B, tensor_dim0=384, tensor_dim1=49,
  //            tile_dim0=384, tile_dim1=49, tensor_dim0_stride=384.
  if (wave == 0) {
    const unsigned lds_addr = (unsigned)(uintptr_t)(smem + QKS_OFF); // flat low32 == LDS byte offset
    const unsigned long long ga = (unsigned long long)(uintptr_t)xw;
    v4u g0;
    g0[0] = 1u;                                       // count=1, user descriptor
    g0[1] = lds_addr;                                 // bits[63:32]
    g0[2] = (unsigned)(ga & 0xffffffffu);             // global_addr[31:0]
    g0[3] = (unsigned)((ga >> 32) & 0x01ffffffu)      // global_addr[56:32]
          | (2u << 30);                               // type=2 ("image")
    v8i g1;
    g1[0] = (int)(2u << 16);                          // data_size=2 -> 4 bytes
    g1[1] = (int)((unsigned)(EMB & 0xffff) << 16);    // tensor_dim0[15:0] @ bits[63:48]
    g1[2] = (int)((unsigned)(N_TOK & 0xffff) << 16);  // dim0 hi=0 | tensor_dim1[15:0]
    g1[3] = (int)((unsigned)(EMB & 0xffff) << 16);    // dim1 hi=0 | tile_dim0
    g1[4] = (int)(N_TOK & 0xffff);                    // tile_dim1=49, tile_dim2=0
    g1[5] = (int)EMB;                                 // tensor_dim0_stride[31:0]
    g1[6] = 0;                                        // stride0 hi | stride1 lo
    g1[7] = 0;
    v4i gz  = {0, 0, 0, 0};                           // 2-D tensor: groups 2/3 unused
    v8i gz8 = {0, 0, 0, 0, 0, 0, 0, 0};               // 6-arg toolchain extra group
    __builtin_amdgcn_tensor_load_to_lds(g0, g1, gz, gz, gz8, 0);
    __builtin_amdgcn_s_wait_tensorcnt(0);
  }
  __syncthreads();
  // -------- Phase 0b: staged f32 -> bf16 xs, rows 49..63 zeroed --------
  {
    const float* xstage = (const float*)(smem + QKS_OFF);
    for (int i4 = tid; i4 < (N_PAD * EMB) / 4; i4 += 256) {
      const int row = i4 / (EMB / 4);
      const int c4  = (i4 % (EMB / 4)) * 4;
      __bf16 v0 = (__bf16)0.f, v1 = v0, v2 = v0, v3 = v0;
      if (row < N_TOK) {
        const float4 f = *(const float4*)(xstage + row * EMB + c4);
        v0 = (__bf16)f.x; v1 = (__bf16)f.y; v2 = (__bf16)f.z; v3 = (__bf16)f.w;
      }
      __bf16* d = xs + row * EMB + c4;
      d[0] = v0; d[1] = v1; d[2] = v2; d[3] = v3;
    }
  }
#else
  // -------- Fallback: direct global->LDS convert --------
  for (int i4 = tid; i4 < (N_PAD * EMB) / 4; i4 += 256) {
    const int row = i4 / (EMB / 4);
    const int c4  = (i4 % (EMB / 4)) * 4;
    __bf16 v0 = (__bf16)0.f, v1 = v0, v2 = v0, v3 = v0;
    if (row < N_TOK) {
      const float4 f = *(const float4*)(xw + row * EMB + c4);
      v0 = (__bf16)f.x; v1 = (__bf16)f.y; v2 = (__bf16)f.z; v3 = (__bf16)f.w;
    }
    __bf16* d = xs + row * EMB + c4;
    d[0] = v0; d[1] = v1; d[2] = v2; d[3] = v3;
  }
#endif
  __syncthreads();   // xs ready; staging region may now be overwritten by qks

  // -------- Phase 1: QKV = X @ Wqkv^T + b  ([64 x 1152]) --------
  // Register-blocked: one B fragment reused across 4 M-tile accumulators.
  for (int nt = wave; nt < 72; nt += 8) {
    const int Nb = nt * 16;
    v8f a0 = {}, a1 = {}, a2 = {}, a3 = {};
#pragma unroll
    for (int kb = 0; kb < 12; ++kb) {
      const int K = kb * 32;
      v16bf bf = load_frag(qkvw, EMB, Nb, K, lane);   // W row j is B column j
      a0 = wmma_bf16(load_frag(xs, EMB,  0, K, lane), bf, a0);
      a1 = wmma_bf16(load_frag(xs, EMB, 16, K, lane), bf, a1);
      a2 = wmma_bf16(load_frag(xs, EMB, 32, K, lane), bf, a2);
      a3 = wmma_bf16(load_frag(xs, EMB, 48, K, lane), bf, a3);
    }
    const float bias = qkvb[Nb + lcol];
    store_qkv_acc(a0,  0, Nb, lane, qks, vts, bias);
    store_qkv_acc(a1, 16, Nb, lane, qks, vts, bias);
    store_qkv_acc(a2, 32, Nb, lane, qks, vts, bias);
    store_qkv_acc(a3, 48, Nb, lane, qks, vts, bias);
  }
  __syncthreads();

  // -------- Phase 2: per-(head, m-tile) attention; waves independent --------
  const float scale = 0.17677669529663687f;   // HD^-0.5
  float*  ssw = ss + wave * 16 * SS_LD;
  __bf16* psw = ps + wave * 16 * PS_LD;
  for (int it = wave; it < N_HEADS * 4; it += 8) {
    const int h  = it >> 2;
    const int mt = it & 3;
    const int Mb = mt * 16;

    // S = Q @ K^T (K=HD=32 -> single WMMA per tile), then scale+bias+mask.
    v16bf qa = load_frag(qks + h * HEAD_D, 768, Mb, 0, lane);
#pragma unroll
    for (int ntile = 0; ntile < 4; ++ntile) {
      v16bf kf = load_frag(qks + 384 + h * HEAD_D, 768, ntile * 16, 0, lane);
      v8f sacc = {};
      sacc = wmma_bf16(qa, kf, sacc);
      const int mcol = ntile * 16 + lcol;
#pragma unroll
      for (int r = 0; r < 8; ++r) {
        const int lrow = lh8 + r;
        const int n = Mb + lrow;
        float v = -1e30f;
        if (n < N_TOK && mcol < N_TOK)
          v = sacc[r] * scale + bias_exp[(h * N_TOK + n) * N_TOK + mcol];
        ssw[lrow * SS_LD + mcol] = v;
      }
    }

    // Row softmax: lane -> row (lane&15), halves split columns, combine via shfl.
    {
      const int row = lcol;
      const float* sr = ssw + row * SS_LD + (lane >> 4) * 32;
      float m = -1e30f;
#pragma unroll
      for (int c = 0; c < 32; ++c) m = fmaxf(m, sr[c]);
      m = fmaxf(m, __shfl_xor(m, 16, 32));
      float ex[32];
      float sum = 0.f;
#pragma unroll
      for (int c = 0; c < 32; ++c) { ex[c] = __expf(sr[c] - m); sum += ex[c]; }
      sum += __shfl_xor(sum, 16, 32);
      const float inv = 1.f / sum;
      __bf16* pr = psw + row * PS_LD + (lane >> 4) * 32;
#pragma unroll
      for (int c = 0; c < 32; ++c) pr[c] = (__bf16)(ex[c] * inv);
    }

    // O = P @ V  (K=64 padded tokens, pad cols of P are exactly 0).
#pragma unroll
    for (int ntile = 0; ntile < 2; ++ntile) {
      v8f o = {};
#pragma unroll
      for (int kt = 0; kt < 2; ++kt) {
        v16bf pa = load_frag(psw, PS_LD, 0, kt * 32, lane);
        v16bf vb = load_frag(vts + (h * HEAD_D + ntile * 16) * 64, 64, 0, kt * 32, lane);
        o = wmma_bf16(pa, vb, o);
      }
      const int dcol = h * HEAD_D + ntile * 16 + lcol;
#pragma unroll
      for (int r = 0; r < 8; ++r) {
        const int n = Mb + lh8 + r;
        xs[n * EMB + dcol] = (n < N_TOK) ? (__bf16)o[r] : (__bf16)0.f;  // zero pad rows
      }
    }
  }
  __syncthreads();

  // -------- Phase 3: out = attn_out @ Wproj^T + b (f32 to HBM) --------
  float* outw = out + (size_t)b * N_TOK * EMB;
  for (int nt = wave; nt < 24; nt += 8) {
    const int Nb = nt * 16;
    v8f a0 = {}, a1 = {}, a2 = {}, a3 = {};
#pragma unroll
    for (int kb = 0; kb < 12; ++kb) {
      const int K = kb * 32;
      v16bf bf = load_frag(projw, EMB, Nb, K, lane);
      a0 = wmma_bf16(load_frag(xs, EMB,  0, K, lane), bf, a0);
      a1 = wmma_bf16(load_frag(xs, EMB, 16, K, lane), bf, a1);
      a2 = wmma_bf16(load_frag(xs, EMB, 32, K, lane), bf, a2);
      a3 = wmma_bf16(load_frag(xs, EMB, 48, K, lane), bf, a3);
    }
    const int j = Nb + lcol;
    const float bias = projb[j];
    v8f accs[4] = {a0, a1, a2, a3};
#pragma unroll
    for (int mt = 0; mt < 4; ++mt) {
#pragma unroll
      for (int r = 0; r < 8; ++r) {
        const int n = mt * 16 + lh8 + r;
        if (n < N_TOK) outw[(size_t)n * EMB + j] = accs[mt][r] + bias;
      }
    }
  }
}

// ---------------- host launch ----------------
extern "C" void kernel_launch(void* const* d_in, const int* in_sizes, int n_in,
                              void* d_out, int out_size, void* d_ws, size_t ws_size,
                              hipStream_t stream) {
  (void)in_sizes; (void)n_in; (void)out_size; (void)ws_size;
  const float* x      = (const float*)d_in[0];
  const float* qkv_w  = (const float*)d_in[1];
  const float* qkv_b  = (const float*)d_in[2];
  const float* proj_w = (const float*)d_in[3];
  const float* proj_b = (const float*)d_in[4];
  const float* rbt    = (const float*)d_in[5];
  float* out = (float*)d_out;

  // workspace: bf16 weights + expanded bias (~1.3 MB)
  char* ws = (char*)d_ws;
  unsigned short* qkvw_bf  = (unsigned short*)(ws);                    // 1152*384
  unsigned short* projw_bf = (unsigned short*)(ws + 884736);           // 384*384
  float*          bias_exp = (float*)(ws + 884736 + 294912);           // 12*49*49

  prep_kernel<<<512, 256, 0, stream>>>(qkv_w, proj_w, rbt, qkvw_bf, projw_bf, bias_exp);

  swin_attn_kernel<<<2048, 256, (size_t)LDS_BYTES, stream>>>(
      x, qkvw_bf, qkv_b, projw_bf, proj_b, bias_exp, out);
}
